// MinimalGPT_69492570849683
// MI455X (gfx1250) — matrix operand order repaired and checked by reading, not compiled, not run on hardware
//
#include <hip/hip_runtime.h>
#include <math.h>

#define B_  2
#define S_  2048
#define D_  512
#define L_  4
#define H_  8
#define HD_ 64
#define V_  32000
#define EPS_ 1e-5f

typedef __bf16 bf16;
typedef __attribute__((ext_vector_type(16))) __bf16 v16bf;
typedef __attribute__((ext_vector_type(8)))  __bf16 v8bf;
typedef __attribute__((ext_vector_type(4)))  __bf16 v4bf;
typedef __attribute__((ext_vector_type(8)))  float  v8f;

__device__ __forceinline__ float gelu_exact(float x) {
    return 0.5f * x * (1.0f + erff(x * 0.70710678118654752f));
}

// Build a 16-bit WMMA fragment from two contiguous 16-byte LDS chunks.
__device__ __forceinline__ v16bf frag2(const bf16* lo, const bf16* hi8) {
    v8bf a = *(const v8bf*)lo;
    v8bf b = *(const v8bf*)hi8;
    return __builtin_shufflevector(a, b, 0, 1, 2, 3, 4, 5, 6, 7,
                                         8, 9, 10, 11, 12, 13, 14, 15);
}

__device__ __forceinline__ v4bf cvt4(float4 f) {
    v4bf r = { (bf16)f.x, (bf16)f.y, (bf16)f.z, (bf16)f.w };
    return r;
}

// ---------------------------------------------------------------- embedding
__global__ void embed_kernel(const int* __restrict__ ids,
                             const float* __restrict__ tok,
                             const float* __restrict__ pos,
                             float* __restrict__ x) {
    int idx4 = blockIdx.x * blockDim.x + threadIdx.x;   // float4 index
    if (idx4 >= B_ * S_ * D_ / 4) return;
    int idx = idx4 * 4;
    int d  = idx & (D_ - 1);
    int bt = idx / D_;
    int t  = bt & (S_ - 1);
    int id = ids[bt];
    float4 a = *(const float4*)(tok + (size_t)id * D_ + d);
    float4 p = *(const float4*)(pos + (size_t)t * D_ + d);
    float4 r = { a.x + p.x, a.y + p.y, a.z + p.z, a.w + p.w };
    *(float4*)(x + idx) = r;
}

// ---------------------------------------------------------------- layernorm
// 256 threads = 8 wave32s; wave-level shfl reductions + tiny LDS combine.
__global__ void ln_kernel(const float* __restrict__ x,
                          const float* __restrict__ w,
                          const float* __restrict__ b,
                          float* __restrict__ y) {
    __shared__ float ws1[8];
    __shared__ float ws2[8];
    int row = blockIdx.x;
    int tid = threadIdx.x;
    int lane = tid & 31;
    int wv   = tid >> 5;
    const float* xr = x + (size_t)row * D_;
    float v0 = xr[tid], v1 = xr[tid + 256];
    float s  = v0 + v1;
    float sq = v0 * v0 + v1 * v1;
    #pragma unroll
    for (int off = 1; off < 32; off <<= 1) {
        s  += __shfl_xor(s,  off, 32);
        sq += __shfl_xor(sq, off, 32);
    }
    if (lane == 0) { ws1[wv] = s; ws2[wv] = sq; }
    __syncthreads();
    float ts = 0.0f, tq = 0.0f;
    #pragma unroll
    for (int i = 0; i < 8; ++i) { ts += ws1[i]; tq += ws2[i]; }
    float mean = ts * (1.0f / D_);
    float var  = tq * (1.0f / D_) - mean * mean;
    float rstd = rsqrtf(var + EPS_);
    float* yr = y + (size_t)row * D_;
    yr[tid]       = (v0 - mean) * rstd * w[tid]       + b[tid];
    yr[tid + 256] = (v1 - mean) * rstd * w[tid + 256] + b[tid + 256];
}

// ---------------------------------------------------------------- WMMA GEMM
// out[M,N] = act(A[M,K] @ W + bias) (+ residual)
// WT=false: W is [K,N]. WT=true: W is [N,K] (tied embedding head).
// Tiles: 64x64x32, 128 threads = 4 waves (2x2), wave tile 32x32 = 2x2 WMMA.
// LDS: A tile K-major [m][k] stride 40; W tile stored TRANSPOSED [n][k]
// stride 40 so both A and B fragments are two ds_load_b128 each.
template<bool GELU_ACT, bool WT>
__global__ void gemm_kernel(const float* __restrict__ A,
                            const float* __restrict__ W,
                            const float* __restrict__ bias,
                            const float* __restrict__ res,
                            float* __restrict__ out,
                            int M, int N, int K) {
    __shared__ bf16 As[64][40];
    __shared__ bf16 Bt[64][40];
    const int tid  = threadIdx.x;
    const int lane = tid & 31;
    const int wid  = tid >> 5;           // 0..3
    const int wm   = (wid >> 1) * 32;
    const int wn   = (wid & 1) * 32;
    const int m0   = blockIdx.y * 64;
    const int n0   = blockIdx.x * 64;
    const int l16  = lane & 15;
    const int hi   = lane >> 4;          // lane half: 0 or 1

    v8f c[2][2] = {};

    for (int k0 = 0; k0 < K; k0 += 32) {
        // ---- stage A (64x32) as bf16, K-major
        #pragma unroll
        for (int e = tid * 4; e < 64 * 32; e += 128 * 4) {
            int i = e >> 5, j = e & 31;
            float4 f = *(const float4*)(A + (size_t)(m0 + i) * K + k0 + j);
            *(v4bf*)&As[i][j] = cvt4(f);
        }
        // ---- stage W (32x64) transposed into Bt[n][k]
        if (WT) {
            #pragma unroll
            for (int e = tid * 4; e < 64 * 32; e += 128 * 4) {
                int n = e >> 5, kk = e & 31;
                float4 f = *(const float4*)(W + (size_t)(n0 + n) * K + k0 + kk);
                *(v4bf*)&Bt[n][kk] = cvt4(f);
            }
        } else {
            #pragma unroll
            for (int e = tid * 4; e < 32 * 64; e += 128 * 4) {
                int kk = e >> 6, n = e & 63;
                float4 f = *(const float4*)(W + (size_t)(k0 + kk) * N + n0 + n);
                Bt[n + 0][kk] = (bf16)f.x;
                Bt[n + 1][kk] = (bf16)f.y;
                Bt[n + 2][kk] = (bf16)f.z;
                Bt[n + 3][kk] = (bf16)f.w;
            }
        }
        // prefetch next K tile while WMMAs run (global_prefetch_b8)
        if (k0 + 32 < K) {
            int pr = tid & 63;
            __builtin_prefetch(A + (size_t)(m0 + pr) * K + (k0 + 32), 0, 3);
            if (WT)
                __builtin_prefetch(W + (size_t)(n0 + pr) * K + (k0 + 32), 0, 3);
            else
                __builtin_prefetch(W + (size_t)(k0 + 32 + (tid & 31)) * N + n0, 0, 3);
        }
        __syncthreads();

        v16bf af[2], wf[2];
        #pragma unroll
        for (int i = 0; i < 2; ++i) {
            const bf16* row = &As[wm + i * 16 + l16][0];
            af[i] = frag2(row + hi * 8, row + 16 + hi * 8);
        }
        #pragma unroll
        for (int j = 0; j < 2; ++j) {
            const bf16* row = &Bt[wn + j * 16 + l16][0];
            wf[j] = frag2(row + hi * 8, row + 16 + hi * 8);
        }
        #pragma unroll
        for (int i = 0; i < 2; ++i)
            #pragma unroll
            for (int j = 0; j < 2; ++j)
                c[i][j] = __builtin_amdgcn_wmma_f32_16x16x32_bf16(
                    false, af[i], false, wf[j], (short)0, c[i][j], false, false);
        __syncthreads();
    }

    // epilogue: bias -> gelu -> residual -> store
    #pragma unroll
    for (int i = 0; i < 2; ++i) {
        #pragma unroll
        for (int j = 0; j < 2; ++j) {
            int n = n0 + wn + j * 16 + l16;
            #pragma unroll
            for (int r = 0; r < 8; ++r) {
                int m = m0 + wm + i * 16 + r + hi * 8;
                float val = c[i][j][r];
                if (bias) val += bias[n];
                if (GELU_ACT) val = gelu_exact(val);
                if (res) val += res[(size_t)m * N + n];
                out[(size_t)m * N + n] = val;
            }
        }
    }
}

// ---------------------------------------------------------------- attention
// Flash-style causal attention, one block per (query-block-of-64, head, batch).
// 128 threads = 4 waves; wave w owns query rows w*16..w*16+15.
// Qs/Ks: [row][d] (K-contiguous for both A- and B-fragments);
// V stored transposed Vt[d][key] so PV B-fragments are contiguous too.
__global__ void attn_kernel(const float* __restrict__ qkv,
                            float* __restrict__ y) {
    __shared__ bf16 Qs[64][72];
    __shared__ bf16 Ks[64][72];
    __shared__ bf16 Vt[64][72];          // [hd][key]
    __shared__ bf16 Ps[4][16][72];       // per-wave P tile [qrow][key]
    const int tid  = threadIdx.x;
    const int lane = tid & 31;
    const int wid  = tid >> 5;
    const int l16  = lane & 15;
    const int hi   = lane >> 4;
    const int qb   = blockIdx.x;         // 0..31
    const int head = blockIdx.y;
    const int b    = blockIdx.z;

    const size_t rs = 3 * D_;
    const float scale = 0.125f;          // 1/sqrt(HD)

    #pragma unroll
    for (int e = tid * 4; e < 64 * 64; e += 128 * 4) {
        int r = e >> 6, d = e & 63;
        int t = qb * 64 + r;
        float4 f = *(const float4*)(qkv + (size_t)(b * S_ + t) * rs + head * HD_ + d);
        v4bf q4 = { (bf16)(f.x * scale), (bf16)(f.y * scale),
                    (bf16)(f.z * scale), (bf16)(f.w * scale) };
        *(v4bf*)&Qs[r][d] = q4;
    }

    float mrow[8], lrow[8];
    v8f o[4] = {};
    #pragma unroll
    for (int r = 0; r < 8; ++r) { mrow[r] = -1e30f; lrow[r] = 0.0f; }

    for (int kb = 0; kb <= qb; ++kb) {
        #pragma unroll
        for (int e = tid * 4; e < 64 * 64; e += 128 * 4) {
            int r = e >> 6, d = e & 63;
            int t = kb * 64 + r;
            size_t base = (size_t)(b * S_ + t) * rs + head * HD_ + d;
            float4 fk = *(const float4*)(qkv + base + D_);
            float4 fv = *(const float4*)(qkv + base + 2 * D_);
            *(v4bf*)&Ks[r][d] = cvt4(fk);
            Vt[d + 0][r] = (bf16)fv.x;   // transpose scatter
            Vt[d + 1][r] = (bf16)fv.y;
            Vt[d + 2][r] = (bf16)fv.z;
            Vt[d + 3][r] = (bf16)fv.w;
        }
        if (kb < qb) {                   // prefetch next K/V block rows
            int t = (kb + 1) * 64 + (tid & 63);
            size_t base = (size_t)(b * S_ + t) * rs + head * HD_;
            __builtin_prefetch(qkv + base + D_,     0, 3);
            __builtin_prefetch(qkv + base + 2 * D_, 0, 3);
        }
        __syncthreads();

        // S = Q K^T : 16 q-rows x 64 keys per wave
        v8f s[4] = {};
        #pragma unroll
        for (int ks = 0; ks < 2; ++ks) {
            const bf16* qrow = &Qs[wid * 16 + l16][ks * 32];
            v16bf aq = frag2(qrow + hi * 8, qrow + 16 + hi * 8);
            v16bf bk[4];
            #pragma unroll
            for (int j = 0; j < 4; ++j) {
                const bf16* krow = &Ks[j * 16 + l16][ks * 32];
                bk[j] = frag2(krow + hi * 8, krow + 16 + hi * 8);
            }
            #pragma unroll
            for (int j = 0; j < 4; ++j)
                s[j] = __builtin_amdgcn_wmma_f32_16x16x32_bf16(
                    false, aq, false, bk[j], (short)0, s[j], false, false);
        }

        if (kb == qb) {   // causal mask on diagonal block
            #pragma unroll
            for (int j = 0; j < 4; ++j) {
                int n = j * 16 + l16;
                #pragma unroll
                for (int r = 0; r < 8; ++r) {
                    int m = wid * 16 + r + hi * 8;
                    if (n > m) s[j][r] = -1e30f;
                }
            }
        }

        // online softmax (row reductions across 16-lane groups)
        #pragma unroll
        for (int r = 0; r < 8; ++r) {
            float mx = fmaxf(fmaxf(s[0][r], s[1][r]), fmaxf(s[2][r], s[3][r]));
            #pragma unroll
            for (int off = 1; off < 16; off <<= 1)
                mx = fmaxf(mx, __shfl_xor(mx, off, 32));
            float mnew = fmaxf(mrow[r], mx);
            float corr = __expf(mrow[r] - mnew);
            float psum = 0.0f;
            #pragma unroll
            for (int j = 0; j < 4; ++j) {
                float p = __expf(s[j][r] - mnew);
                s[j][r] = p;
                psum += p;
            }
            #pragma unroll
            for (int off = 1; off < 16; off <<= 1)
                psum += __shfl_xor(psum, off, 32);
            lrow[r] = lrow[r] * corr + psum;
            mrow[r] = mnew;
            #pragma unroll
            for (int j2 = 0; j2 < 4; ++j2)
                o[j2][r] = o[j2][r] * corr;
        }

        // P: C-layout fragments -> row-major LDS (per-wave slab)
        #pragma unroll
        for (int j = 0; j < 4; ++j) {
            int n = j * 16 + l16;
            #pragma unroll
            for (int r = 0; r < 8; ++r)
                Ps[wid][r + hi * 8][n] = (bf16)s[j][r];
        }
        __syncthreads();

        // O += P V
        #pragma unroll
        for (int ks = 0; ks < 2; ++ks) {
            const bf16* prow = &Ps[wid][l16][ks * 32];
            v16bf ap = frag2(prow + hi * 8, prow + 16 + hi * 8);
            v16bf bv[4];
            #pragma unroll
            for (int j2 = 0; j2 < 4; ++j2) {
                const bf16* vrow = &Vt[j2 * 16 + l16][ks * 32];  // [hd][key]
                bv[j2] = frag2(vrow + hi * 8, vrow + 16 + hi * 8);
            }
            #pragma unroll
            for (int j2 = 0; j2 < 4; ++j2)
                o[j2] = __builtin_amdgcn_wmma_f32_16x16x32_bf16(
                    false, ap, false, bv[j2], (short)0, o[j2], false, false);
        }
        __syncthreads();
    }

    #pragma unroll
    for (int j2 = 0; j2 < 4; ++j2) {
        int d = head * HD_ + j2 * 16 + l16;
        #pragma unroll
        for (int r = 0; r < 8; ++r) {
            int t = qb * 64 + wid * 16 + r + hi * 8;
            y[(size_t)(b * S_ + t) * D_ + d] = o[j2][r] / lrow[r];
        }
    }
}

// ---------------------------------------------------------------- launch
extern "C" void kernel_launch(void* const* d_in, const int* in_sizes, int n_in,
                              void* d_out, int out_size, void* d_ws, size_t ws_size,
                              hipStream_t stream) {
    (void)in_sizes; (void)n_in; (void)out_size; (void)ws_size;
    const int*   ids     = (const int*)  d_in[0];
    const float* tok_emb = (const float*)d_in[1];
    const float* pos_emb = (const float*)d_in[2];
    const float* ln1_w   = (const float*)d_in[3];
    const float* ln1_b   = (const float*)d_in[4];
    const float* qkv_w   = (const float*)d_in[5];
    const float* qkv_b   = (const float*)d_in[6];
    const float* proj_w  = (const float*)d_in[7];
    const float* proj_b  = (const float*)d_in[8];
    const float* ln2_w   = (const float*)d_in[9];
    const float* ln2_b   = (const float*)d_in[10];
    const float* fc1_w   = (const float*)d_in[11];
    const float* fc1_b   = (const float*)d_in[12];
    const float* fc2_w   = (const float*)d_in[13];
    const float* fc2_b   = (const float*)d_in[14];
    const float* lnf_w   = (const float*)d_in[15];
    const float* lnf_b   = (const float*)d_in[16];
    float* out = (float*)d_out;

    const int M = B_ * S_;                       // 4096 token rows
    float* x    = (float*)d_ws;                  // [M, D]
    float* h    = x    + (size_t)M * D_;         // [M, D]
    float* qkvb = h    + (size_t)M * D_;         // [M, 3D]
    float* yb   = qkvb + (size_t)M * 3 * D_;     // [M, D]
    float* h1   = yb   + (size_t)M * D_;         // [M, 4D]

    dim3 blk128(128);
    embed_kernel<<<(M * D_ / 4 + 255) / 256, 256, 0, stream>>>(ids, tok_emb, pos_emb, x);

    for (int l = 0; l < L_; ++l) {
        ln_kernel<<<M, 256, 0, stream>>>(x, ln1_w + l * D_, ln1_b + l * D_, h);
        gemm_kernel<false, false><<<dim3(3 * D_ / 64, M / 64), blk128, 0, stream>>>(
            h, qkv_w + (size_t)l * D_ * 3 * D_, qkv_b + l * 3 * D_, nullptr, qkvb,
            M, 3 * D_, D_);
        attn_kernel<<<dim3(S_ / 64, H_, B_), blk128, 0, stream>>>(qkvb, yb);
        gemm_kernel<false, false><<<dim3(D_ / 64, M / 64), blk128, 0, stream>>>(
            yb, proj_w + (size_t)l * D_ * D_, proj_b + l * D_, x, x,
            M, D_, D_);
        ln_kernel<<<M, 256, 0, stream>>>(x, ln2_w + l * D_, ln2_b + l * D_, h);
        gemm_kernel<true, false><<<dim3(4 * D_ / 64, M / 64), blk128, 0, stream>>>(
            h, fc1_w + (size_t)l * D_ * 4 * D_, fc1_b + l * 4 * D_, nullptr, h1,
            M, 4 * D_, D_);
        gemm_kernel<false, false><<<dim3(D_ / 64, M / 64), blk128, 0, stream>>>(
            h1, fc2_w + (size_t)l * 4 * D_ * D_, fc2_b + l * D_, x, x,
            M, D_, 4 * D_);
    }
    ln_kernel<<<M, 256, 0, stream>>>(x, lnf_w, lnf_b, h);
    gemm_kernel<false, true><<<dim3(V_ / 64, M / 64), blk128, 0, stream>>>(
        h, tok_emb, nullptr, nullptr, out, M, V_, D_);
}
